// TransformerEncoderBlock_7413113553431
// MI455X (gfx1250) — matrix operand order repaired and checked by reading, not compiled, not run on hardware
//
#include <hip/hip_runtime.h>
#include <math.h>

#define Dm   1024
#define Hn   16
#define HDm  64
#define FFm  4096
#define Bm   2
#define Sm   2048
#define Mrows (Bm*Sm)          // 4096 token rows
#define NEGV (-9000000000000000.0f)
#define LN_EPS 1e-5f

typedef __attribute__((ext_vector_type(16))) __bf16 v16bf;
typedef __attribute__((ext_vector_type(8)))  __bf16 v8bf;
typedef __attribute__((ext_vector_type(8)))  float  v8f;

// ---------------------------------------------------------------------------
// WMMA fragment loaders (CDNA5 16x16x32 bf16 layouts, CDNA5 ISA)
// A: 16x32 (MxK), lane<16: row=lane, K = {0..7,16..23}; lane>=16: K+=8.
// B: 32x16 (KxN), col=lane&15, K = {0..7,16..23}; lane>=16: K+=8.
// C/D: elem r -> row = r + (lane>=16?8:0), col = lane&15.
// ---------------------------------------------------------------------------
__device__ __forceinline__ v16bf frag_a_load(const __bf16* base, int lda, int lane) {
  int row = lane & 15;
  int kb  = (lane >> 4) << 3;          // 0 or 8
  const __bf16* p = base + row * lda + kb;
  v8bf lo = *(const v8bf*)p;           // K = kb .. kb+7
  v8bf hi = *(const v8bf*)(p + 16);    // K = kb+16 .. kb+23
  v16bf r;
#pragma unroll
  for (int i = 0; i < 8; ++i) { r[i] = lo[i]; r[i + 8] = hi[i]; }
  return r;
}

__device__ __forceinline__ v16bf frag_b_load(const __bf16* base, int ldb, int lane) {
  int col = lane & 15;
  int kb  = (lane >> 4) << 3;          // 0 or 8
  const __bf16* p = base + col * ldb + kb;
  v8bf lo = *(const v8bf*)p;           // K = kb .. kb+7
  v8bf hi = *(const v8bf*)(p + 16);    // K = kb+16 .. kb+23
  v16bf r;
#pragma unroll
  for (int i = 0; i < 8; ++i) { r[i] = lo[i]; r[i + 8] = hi[i]; }
  return r;
}

__device__ __forceinline__ v8f wmma_bf16(v16bf a, v16bf b, v8f c) {
  return __builtin_amdgcn_wmma_f32_16x16x32_bf16(false, a, false, b,
                                                 (short)0, c, false, false);
}

// ---------------------------------------------------------------------------
// Weight convert+transpose: W[K,N] f32 -> WT[N,K] bf16
// ---------------------------------------------------------------------------
__global__ void wt_kernel(const float* __restrict__ W, __bf16* __restrict__ WT,
                          int K, int N) {
  size_t idx = (size_t)blockIdx.x * 256 + threadIdx.x;
  int n = (int)(idx / K);
  int k = (int)(idx % K);
  WT[idx] = (__bf16)W[(size_t)k * N + n];
}

// ---------------------------------------------------------------------------
// LayerNorm over D=1024, one block (256 thr) per row; writes f32 + bf16
// ---------------------------------------------------------------------------
__global__ void ln_kernel(const float* __restrict__ x, const float* __restrict__ g,
                          const float* __restrict__ be, float* __restrict__ outF,
                          __bf16* __restrict__ outB) {
  __shared__ float sh1[256], sh2[256];
  int row = blockIdx.x;
  const float* xr = x + (size_t)row * Dm;
  float v[4], s = 0.f, ss = 0.f;
#pragma unroll
  for (int i = 0; i < 4; ++i) {
    v[i] = xr[threadIdx.x + 256 * i];
    s += v[i]; ss += v[i] * v[i];
  }
  sh1[threadIdx.x] = s; sh2[threadIdx.x] = ss;
  __syncthreads();
  for (int st = 128; st > 0; st >>= 1) {
    if (threadIdx.x < st) {
      sh1[threadIdx.x] += sh1[threadIdx.x + st];
      sh2[threadIdx.x] += sh2[threadIdx.x + st];
    }
    __syncthreads();
  }
  float mu  = sh1[0] * (1.0f / Dm);
  float var = sh2[0] * (1.0f / Dm) - mu * mu;
  float rs  = rsqrtf(var + LN_EPS);
#pragma unroll
  for (int i = 0; i < 4; ++i) {
    int c = threadIdx.x + 256 * i;
    float o = (v[i] - mu) * rs * g[c] + be[c];
    outF[(size_t)row * Dm + c] = o;
    outB[(size_t)row * Dm + c] = (__bf16)o;
  }
}

// ---------------------------------------------------------------------------
// Generic bf16 WMMA GEMM: C[M,N] = A[M,K] @ BT[N,K]^T, one 16x16 tile / wave.
// MODE 0: +bias -> bf16     MODE 1: +bias +res -> f32     MODE 2: +bias, GELU -> bf16
// ---------------------------------------------------------------------------
template <int MODE>
__global__ __launch_bounds__(256, 1)
void gemm_kernel(const __bf16* __restrict__ A, const __bf16* __restrict__ BT,
                 const float* __restrict__ bias, const float* __restrict__ res,
                 float* __restrict__ outF, __bf16* __restrict__ outB,
                 int N, int K) {
  int lane = threadIdx.x & 31;
  int wave = threadIdx.x >> 5;
  int tile = blockIdx.x * 8 + wave;
  int tilesN = N >> 4;
  int tm = tile / tilesN;
  int tn = tile % tilesN;
  const __bf16* Ab = A  + (size_t)tm * 16 * K;
  const __bf16* Bb = BT + (size_t)tn * 16 * K;
  v8f acc = {};
  for (int k0 = 0; k0 < K; k0 += 32) {
    if (k0 + 32 < K) {                       // gfx1250 global_prefetch_b8
      __builtin_prefetch(Ab + k0 + 32, 0, 1);
      __builtin_prefetch(Bb + k0 + 32, 0, 1);
    }
    v16bf a = frag_a_load(Ab + k0, K, lane);
    v16bf b = frag_b_load(Bb + k0, K, lane);
    acc = wmma_bf16(a, b, acc);
  }
  int roff = (lane >> 4) << 3;
  int col  = lane & 15;
  int n    = tn * 16 + col;
  float bn = bias[n];
#pragma unroll
  for (int r = 0; r < 8; ++r) {
    int row = tm * 16 + roff + r;
    float v = acc[r] + bn;
    size_t o = (size_t)row * N + n;
    if (MODE == 0) {
      outB[o] = (__bf16)v;
    } else if (MODE == 1) {
      outF[o] = v + res[o];
    } else {
      float ge = 0.5f * v * (1.0f + erff(v * 0.70710678118f));
      outB[o] = (__bf16)ge;
    }
  }
}

// ---------------------------------------------------------------------------
// RoPE + repack: qkv[M, H*3*HD] bf16 -> q,k rotated [B,H,S,HD] bf16, vT [B,H,HD,S]
// ---------------------------------------------------------------------------
__global__ void rope_kernel(const __bf16* __restrict__ qkv, const float* __restrict__ sp,
                            __bf16* __restrict__ qro, __bf16* __restrict__ kro,
                            __bf16* __restrict__ vT) {
  size_t idx = (size_t)blockIdx.x * 256 + threadIdx.x;   // over B*S*H*HD
  int j = (int)(idx % HDm);
  int h = (int)((idx / HDm) % Hn);
  int s = (int)((idx / (HDm * Hn)) % Sm);
  int b = (int)(idx / ((size_t)HDm * Hn * Sm));
  const __bf16* row = qkv + (((size_t)b * Sm + s) * Hn + h) * (3 * HDm);
  float q = (float)row[j];
  float k = (float)row[HDm + j];
  float v = (float)row[2 * HDm + j];
  int   jr   = (j < 32) ? (j + 32) : (j - 32);
  float sign = (j < 32) ? -1.f : 1.f;
  float qr = sign * (float)row[jr];
  float kr = sign * (float)row[HDm + jr];
  float a  = sp[(size_t)s * HDm + j];
  float c  = cosf(a), sn = sinf(a);
  size_t bh = (size_t)b * Hn + h;
  qro[(bh * Sm + s) * HDm + j] = (__bf16)(q * c + qr * sn);
  kro[(bh * Sm + s) * HDm + j] = (__bf16)(k * c + kr * sn);
  vT[(bh * HDm + j) * Sm + s]  = (__bf16)v;
}

// ---------------------------------------------------------------------------
// Flash attention: one wave per 16-row query tile; stream 32 keys per step.
// S = Q@K^T (4 WMMA), +pos_bias(NT loads), *1/sqrt(HD), mask, online softmax,
// P relaid via per-wave LDS tile, O += P@V (4 WMMA).
// __launch_bounds__(256,1): keep Q/O/S fragments register-resident (no spill).
// ---------------------------------------------------------------------------
__global__ __launch_bounds__(256, 1)
void flash_kernel(const __bf16* __restrict__ qro, const __bf16* __restrict__ kro,
                  const __bf16* __restrict__ vT, const float* __restrict__ pos_bias,
                  const int* __restrict__ mask, __bf16* __restrict__ valsb) {
  __shared__ __bf16 plds[8][16 * 32];
  int lane = threadIdx.x & 31;
  int wave = threadIdx.x >> 5;
  int tile = blockIdx.x * 8 + wave;      // 0 .. B*H*(S/16)-1
  int q0 = (tile % (Sm / 16)) * 16;
  int bh = tile / (Sm / 16);
  int h  = bh % Hn;
  int b  = bh / Hn;

  const __bf16* qbase = qro + ((size_t)bh * Sm + q0) * HDm;
  v16bf qa0 = frag_a_load(qbase,      HDm, lane);   // K 0..31 of head dim
  v16bf qa1 = frag_a_load(qbase + 32, HDm, lane);   // K 32..63

  v8f o0 = {}, o1 = {}, o2 = {}, o3 = {};
  float m[8], l[8];
#pragma unroll
  for (int r = 0; r < 8; ++r) { m[r] = -3.0e38f; l[r] = 0.f; }

  int roff = (lane >> 4) << 3;
  int col  = lane & 15;

  for (int j0 = 0; j0 < Sm; j0 += 32) {
    const __bf16* kbase = kro + ((size_t)bh * Sm + j0) * HDm;
    v8f s0 = {}, s1 = {};
    s0 = wmma_bf16(qa0, frag_b_load(kbase,            HDm, lane), s0);
    s0 = wmma_bf16(qa1, frag_b_load(kbase + 32,       HDm, lane), s0);
    s1 = wmma_bf16(qa0, frag_b_load(kbase + 16 * HDm,      HDm, lane), s1);
    s1 = wmma_bf16(qa1, frag_b_load(kbase + 16 * HDm + 32, HDm, lane), s1);

#pragma unroll
    for (int r = 0; r < 8; ++r) {
      int qrow = q0 + roff + r;
      int c0 = j0 + col, c1 = j0 + 16 + col;
      // pos_bias: 256MB streamed once -> non-temporal (don't pollute 192MB L2)
      float pb0 = __builtin_nontemporal_load(&pos_bias[((size_t)h * Sm + qrow) * Sm + c0]);
      float pb1 = __builtin_nontemporal_load(&pos_bias[((size_t)h * Sm + qrow) * Sm + c1]);
      float t0 = (s0[r] + pb0) * 0.125f;
      float t1 = (s1[r] + pb1) * 0.125f;
      t0 = (mask[(size_t)qrow * Sm + c0] == 0) ? NEGV : t0;
      t1 = (mask[(size_t)qrow * Sm + c1] == 0) ? NEGV : t1;
      // row-wise max across the 16 lanes of this half-wave
      float rm = fmaxf(t0, t1);
#pragma unroll
      for (int d = 1; d < 16; d <<= 1) rm = fmaxf(rm, __shfl_xor(rm, d, 32));
      float mn = fmaxf(m[r], rm);
      float al = __expf(m[r] - mn);
      m[r] = mn;
      float p0 = __expf(t0 - mn);
      float p1 = __expf(t1 - mn);
      float rs = p0 + p1;
#pragma unroll
      for (int d = 1; d < 16; d <<= 1) rs += __shfl_xor(rs, d, 32);
      l[r] = l[r] * al + rs;
      o0[r] *= al; o1[r] *= al; o2[r] *= al; o3[r] *= al;
      plds[wave][(roff + r) * 32 + col]      = (__bf16)p0;
      plds[wave][(roff + r) * 32 + col + 16] = (__bf16)p1;
    }
    asm volatile("s_wait_dscnt 0" ::: "memory");   // wave-local LDS RAW fence
    v16bf pf = frag_a_load(&plds[wave][0], 32, lane);

    const __bf16* vbase = vT + (size_t)bh * HDm * Sm + j0;
    o0 = wmma_bf16(pf, frag_b_load(vbase,           Sm, lane), o0);
    o1 = wmma_bf16(pf, frag_b_load(vbase + 16 * Sm, Sm, lane), o1);
    o2 = wmma_bf16(pf, frag_b_load(vbase + 32 * Sm, Sm, lane), o2);
    o3 = wmma_bf16(pf, frag_b_load(vbase + 48 * Sm, Sm, lane), o3);
  }

#pragma unroll
  for (int r = 0; r < 8; ++r) {
    int qrow = q0 + roff + r;
    float inv = 1.0f / l[r];
    size_t base = ((size_t)b * Sm + qrow) * Dm + (size_t)h * HDm;
    valsb[base + col]      = (__bf16)(o0[r] * inv);
    valsb[base + 16 + col] = (__bf16)(o1[r] * inv);
    valsb[base + 32 + col] = (__bf16)(o2[r] * inv);
    valsb[base + 48 + col] = (__bf16)(o3[r] * inv);
  }
}

// ---------------------------------------------------------------------------
extern "C" void kernel_launch(void* const* d_in, const int* in_sizes, int n_in,
                              void* d_out, int out_size, void* d_ws, size_t ws_size,
                              hipStream_t stream) {
  const float* x        = (const float*)d_in[0];
  const int*   mask     = (const int*)  d_in[1];
  const float* pos_bias = (const float*)d_in[2];
  const float* sp       = (const float*)d_in[3];
  const float* Wqkv     = (const float*)d_in[4];
  const float* bqkv     = (const float*)d_in[5];
  const float* Wo       = (const float*)d_in[6];
  const float* bo       = (const float*)d_in[7];
  const float* W1       = (const float*)d_in[8];
  const float* b1       = (const float*)d_in[9];
  const float* W2       = (const float*)d_in[10];
  const float* b2       = (const float*)d_in[11];
  const float* g1       = (const float*)d_in[12];
  const float* be1      = (const float*)d_in[13];
  const float* g2       = (const float*)d_in[14];
  const float* be2      = (const float*)d_in[15];
  float* out = (float*)d_out;

  char* p = (char*)d_ws;
  auto alloc = [&](size_t bytes) -> char* {
    char* r = p; p += (bytes + 255) & ~(size_t)255; return r;
  };
  __bf16* WqkvT = (__bf16*)alloc((size_t)3 * Dm * Dm * 2);
  __bf16* WoT   = (__bf16*)alloc((size_t)Dm * Dm * 2);
  __bf16* W1T   = (__bf16*)alloc((size_t)FFm * Dm * 2);
  __bf16* W2T   = (__bf16*)alloc((size_t)Dm * FFm * 2);
  float*  xn1f  = (float*) alloc((size_t)Mrows * Dm * 4);
  __bf16* xn1b  = (__bf16*)alloc((size_t)Mrows * Dm * 2);
  __bf16* qkvb  = (__bf16*)alloc((size_t)Mrows * 3 * Dm * 2);
  __bf16* qro   = (__bf16*)alloc((size_t)Bm * Hn * Sm * HDm * 2);
  __bf16* kro   = (__bf16*)alloc((size_t)Bm * Hn * Sm * HDm * 2);
  __bf16* vT    = (__bf16*)alloc((size_t)Bm * Hn * HDm * Sm * 2);
  __bf16* valsb = (__bf16*)alloc((size_t)Mrows * Dm * 2);
  float*  yf    = (float*) alloc((size_t)Mrows * Dm * 4);
  float*  xn2f  = (float*) alloc((size_t)Mrows * Dm * 4);
  __bf16* xn2b  = (__bf16*)alloc((size_t)Mrows * Dm * 2);
  __bf16* hb    = (__bf16*)alloc((size_t)Mrows * FFm * 2);

  // 1) weight conversion / transposition to bf16 [N,K]
  wt_kernel<<<(3 * Dm * Dm) / 256, 256, 0, stream>>>(Wqkv, WqkvT, Dm, 3 * Dm);
  wt_kernel<<<(Dm * Dm) / 256, 256, 0, stream>>>(Wo, WoT, Dm, Dm);
  wt_kernel<<<(Dm * FFm) / 256, 256, 0, stream>>>(W1, W1T, Dm, FFm);
  wt_kernel<<<(FFm * Dm) / 256, 256, 0, stream>>>(W2, W2T, FFm, Dm);

  // 2) LN1
  ln_kernel<<<Mrows, 256, 0, stream>>>(x, g1, be1, xn1f, xn1b);

  // 3) QKV = xn1 @ Wqkv + bqkv   -> bf16 [M, 3D]
  gemm_kernel<0><<<(Mrows / 16) * (3 * Dm / 16) / 8, 256, 0, stream>>>(
      xn1b, WqkvT, bqkv, nullptr, nullptr, qkvb, 3 * Dm, Dm);

  // 4) RoPE + per-head repack (q,k rotated; v transposed)
  rope_kernel<<<((size_t)Bm * Sm * Hn * HDm) / 256, 256, 0, stream>>>(
      qkvb, sp, qro, kro, vT);

  // 5) flash attention -> valsb bf16 [M, D]
  flash_kernel<<<(Bm * Hn * (Sm / 16)) / 8, 256, 0, stream>>>(
      qro, kro, vT, pos_bias, mask, valsb);

  // 6) y = xn1 + (vals @ Wo + bo)  -> f32
  gemm_kernel<1><<<(Mrows / 16) * (Dm / 16) / 8, 256, 0, stream>>>(
      valsb, WoT, bo, xn1f, yf, nullptr, Dm, Dm);

  // 7) LN2
  ln_kernel<<<Mrows, 256, 0, stream>>>(yf, g2, be2, xn2f, xn2b);

  // 8) h = gelu(xn2 @ W1 + b1)  -> bf16 [M, FF]
  gemm_kernel<2><<<(Mrows / 16) * (FFm / 16) / 8, 256, 0, stream>>>(
      xn2b, W1T, b1, nullptr, nullptr, hb, FFm, Dm);

  // 9) out = xn2 + (h @ W2 + b2)  -> f32 d_out
  gemm_kernel<1><<<(Mrows / 16) * (Dm / 16) / 8, 256, 0, stream>>>(
      hb, W2T, b2, xn2f, out, nullptr, Dm, FFm);

  (void)in_sizes; (void)n_in; (void)out_size; (void)ws_size;
}
